// RBM_19524921327712
// MI455X (gfx1250) — matrix operand order repaired and checked by reading, not compiled, not run on hardware
//
#include <hip/hip_runtime.h>
#include <hip/hip_bf16.h>

// ---------------------------------------------------------------------------
// RBM CD-k on MI455X (gfx1250, wave32, WMMA f16->f32)
//
// Chain GEMMs: C = A[MxK] * B[NxK]^T   ("NT": both operands K-contiguous)
//   GEMM-H: H = V  @ Wf^T   (K=NV=4096, N=NH=2048)   Wf  = f16(w)    [NH,NV]
//   GEMM-V: V = H  @ WT^T   (K=NH=2048, N=NV=4096)   WT  = f16(w)^T  [NV,NH]
// Wf (16MB f16) and WT (16MB) stay resident in the 192MB L2 -> compute bound.
// ---------------------------------------------------------------------------

typedef __attribute__((ext_vector_type(16))) _Float16 v16h;
typedef __attribute__((ext_vector_type(8)))  _Float16 v8h;
typedef __attribute__((ext_vector_type(8)))  float    v8f;

#define T_DIM   1024
#define NV_DIM  4096
#define NH_DIM  2048
#define K_STEPS 25

// ---- stateless counter-based RNG (deterministic across graph replays) -----
__device__ __forceinline__ float hash_uniform(unsigned seed, unsigned idx) {
  unsigned s = seed * 0x9E3779B9u ^ (idx * 0x85EBCA6Bu + 0x27220A95u);
  s ^= s >> 16; s *= 0x7FEB352Du;
  s ^= s >> 15; s *= 0x846CA68Bu;
  s ^= s >> 16;
  return (float)(s >> 8) * (1.0f / 16777216.0f);
}

__device__ __forceinline__ float softplus_f(float v) {
  // numerically stable log(1+exp(v))
  return v > 0.0f ? v + log1pf(__expf(-v)) : log1pf(__expf(v));
}

// ---- WMMA fragment loads (layouts per CDNA5 ISA §7.12.2) -------
// A 16x32 f16 (MxK): lane m(0..15) rows; lanes 0-15 hold K {0..7,16..23},
// lanes 16-31 hold K {8..15,24..31}. Rows of A are K-contiguous -> 2x b128.
union AFrag { v16h v; v8h h[2]; };

__device__ __forceinline__ v16h load_a_frag(const _Float16* __restrict__ A,
                                            int row0, int K, int kk, int lane) {
  const int row = row0 + (lane & 15);
  const int hi  = lane >> 4;                 // 0 or 1
  const _Float16* p = A + (size_t)row * K + kk + hi * 8;
  AFrag f;
  f.h[0] = *(const v8h*)(p);                 // K = hi*8 .. hi*8+7
  f.h[1] = *(const v8h*)(p + 16);            // K = 16+hi*8 ..
  return f.v;
}

// B 32x16 f16 (KxN), source row-major [N,K] (row n = column n of B):
// lanes 0-15 hold K {0..7,16..23} of col n, lanes 16-31 hold K {8..15,24..31}.
// Same K order as A -> 2x b128 loads.
__device__ __forceinline__ v16h load_b_frag(const _Float16* __restrict__ B,
                                            int col0, int K, int kk, int lane) {
  const int n  = col0 + (lane & 15);
  const int hi = lane >> 4;
  const _Float16* p = B + (size_t)n * K + kk + hi * 8;
  AFrag f;
  f.h[0] = *(const v8h*)(p);
  f.h[1] = *(const v8h*)(p + 16);
  return f.v;
}

// ---------------------------------------------------------------------------
// Fused GEMM + epilogue.
//   mode 0: logits -> sigmoid -> Bernoulli(hash) -> Out (f16 {0,1})
//   mode 1: per-row sum of -softplus(logit) -> atomicAdd into Fdst[row]
// Wave tile 32Mx64N (2x4 WMMA accum tiles), 4 waves stacked in M => 128Mx64N.
// ---------------------------------------------------------------------------
__global__ __launch_bounds__(128)
void rbm_gemm_nt(const _Float16* __restrict__ A, const _Float16* __restrict__ B,
                 const float* __restrict__ bias, _Float16* __restrict__ Out,
                 float* __restrict__ Fdst, int Kdim, int Ndim, int mode,
                 unsigned seed) {
  const int lane = threadIdx.x & 31;
  const int wave = threadIdx.x >> 5;
  const int t0   = blockIdx.x * 128 + wave * 32;
  const int n0   = blockIdx.y * 64;

  v8f c[2][4] = {};

  for (int kk = 0; kk < Kdim; kk += 32) {
    v16h a0 = load_a_frag(A, t0,      Kdim, kk, lane);
    v16h a1 = load_a_frag(A, t0 + 16, Kdim, kk, lane);
#pragma unroll
    for (int j = 0; j < 4; ++j) {
      v16h b = load_b_frag(B, n0 + j * 16, Kdim, kk, lane);
      c[0][j] = __builtin_amdgcn_wmma_f32_16x16x32_f16(
          false, a0, false, b, (short)0, c[0][j], false, false);
      c[1][j] = __builtin_amdgcn_wmma_f32_16x16x32_f16(
          false, a1, false, b, (short)0, c[1][j], false, false);
    }
  }

  // C/D layout: lanes 0-15 -> (M = vgpr r, N = lane); lanes 16-31 -> M = r+8.
  const int nlane = lane & 15;
  const int mhi   = (lane >> 4) * 8;

  float bb[4];
#pragma unroll
  for (int j = 0; j < 4; ++j) bb[j] = bias[n0 + j * 16 + nlane];

  if (mode == 0) {
#pragma unroll
    for (int i = 0; i < 2; ++i)
#pragma unroll
      for (int j = 0; j < 4; ++j)
#pragma unroll
        for (int r = 0; r < 8; ++r) {
          const int m = t0 + i * 16 + mhi + r;
          const int n = n0 + j * 16 + nlane;
          const float logit = c[i][j][r] + bb[j];
          const float p = 1.0f / (1.0f + __expf(-logit));
          const float u = hash_uniform(seed, (unsigned)m * (unsigned)Ndim + n);
          Out[(size_t)m * Ndim + n] = (_Float16)((u < p) ? 1.0f : 0.0f);
        }
  } else {
#pragma unroll
    for (int i = 0; i < 2; ++i)
#pragma unroll
      for (int r = 0; r < 8; ++r) {
        float s = 0.0f;
#pragma unroll
        for (int j = 0; j < 4; ++j) s -= softplus_f(c[i][j][r] + bb[j]);
        // reduce the 16 lanes that share this output row
#pragma unroll
        for (int mk = 1; mk < 16; mk <<= 1) s += __shfl_xor(s, mk, 32);
        if (nlane == 0)
          atomicAdd(&Fdst[t0 + i * 16 + mhi + r], s);
      }
  }
}

// ---- one-time: w (f32) -> Wf (f16 [NH,NV]) and WT (f16 [NV,NH]) -----------
__global__ __launch_bounds__(256)
void convert_transpose_w(const float* __restrict__ w,
                         _Float16* __restrict__ Wf,
                         _Float16* __restrict__ WT) {
  __shared__ _Float16 tile[32][33];
  const int kv0 = blockIdx.x * 32;
  const int h0  = blockIdx.y * 32;
  const int tx  = threadIdx.x;   // 0..31
  const int ty  = threadIdx.y;   // 0..7
#pragma unroll
  for (int i = 0; i < 32; i += 8) {
    const int h  = h0 + ty + i;
    const int kv = kv0 + tx;
    const _Float16 hv = (_Float16)w[(size_t)h * NV_DIM + kv];
    Wf[(size_t)h * NV_DIM + kv] = hv;
    tile[ty + i][tx] = hv;
  }
  __syncthreads();
#pragma unroll
  for (int i = 0; i < 32; i += 8) {
    const int kv = kv0 + ty + i;
    const int h  = h0 + tx;
    WT[(size_t)kv * NH_DIM + h] = tile[tx][ty + i];
  }
}

// ---- one-time: x (f32 binary) -> Vx (kept) and Vcur (chain state), f16 ----
__global__ __launch_bounds__(256)
void convert_x(const float* __restrict__ x, _Float16* __restrict__ Vx,
               _Float16* __restrict__ Vc, int n) {
  const int i = blockIdx.x * blockDim.x + threadIdx.x;
  if (i < n) {
    const _Float16 h = (_Float16)x[i];
    Vx[i] = h;
    Vc[i] = h;
  }
}

__global__ void init_zero(float* __restrict__ Fx, float* __restrict__ Fv,
                          float* __restrict__ out) {
  const int i = blockIdx.x * blockDim.x + threadIdx.x;
  if (i < T_DIM) { Fx[i] = 0.0f; Fv[i] = 0.0f; }
  if (i == 0) out[0] = 0.0f;
}

// ---- finalize: cost = sum_t [ (Fx[t] - x.bv) - (Fv[t] - vo.bv) ] / T -------
__global__ __launch_bounds__(256)
void rbm_finalize(const float* __restrict__ x, const _Float16* __restrict__ vo,
                  const float* __restrict__ bv, const float* __restrict__ Fx,
                  const float* __restrict__ Fv, float* __restrict__ out) {
  __shared__ float sdata[256];
  const int t = blockIdx.x;
  const size_t base = (size_t)t * NV_DIM;
  float acc = 0.0f;
  for (int k = threadIdx.x; k < NV_DIM; k += 256) {
    const float b = bv[k];
    acc += ((float)vo[base + k] - x[base + k]) * b;   // dot(vo,bv) - dot(x,bv)
  }
  sdata[threadIdx.x] = acc;
  __syncthreads();
  for (int s = 128; s > 0; s >>= 1) {
    if (threadIdx.x < s) sdata[threadIdx.x] += sdata[threadIdx.x + s];
    __syncthreads();
  }
  if (threadIdx.x == 0) {
    // F(x)-F(vo) = (Fx - x.bv) - (Fv - vo.bv) = Fx - Fv + (vo.bv - x.bv)
    const float contrib = Fx[t] - Fv[t] + sdata[0];
    atomicAdd(out, contrib * (1.0f / (float)T_DIM));
  }
}

// ---------------------------------------------------------------------------
extern "C" void kernel_launch(void* const* d_in, const int* in_sizes, int n_in,
                              void* d_out, int out_size, void* d_ws, size_t ws_size,
                              hipStream_t stream) {
  const float* x  = (const float*)d_in[0];   // [T, NV] binary
  const float* w  = (const float*)d_in[1];   // [NH, NV]
  const float* bh = (const float*)d_in[2];   // [NH]
  const float* bv = (const float*)d_in[3];   // [NV]
  float* out = (float*)d_out;                // scalar cost

  // workspace layout (hipMalloc'd base => 256B aligned; offsets 1MB mult except Fv (+8KB))
  char* ws = (char*)d_ws;
  const size_t MB = 1024 * 1024;
  _Float16* Wf = (_Float16*)(ws + 0 * MB);    // [NH,NV] f16  16MB
  _Float16* WT = (_Float16*)(ws + 16 * MB);   // [NV,NH] f16  16MB
  _Float16* Vx = (_Float16*)(ws + 32 * MB);   // [T,NV]  f16   8MB (kept = x)
  _Float16* Vc = (_Float16*)(ws + 40 * MB);   // [T,NV]  f16   8MB (chain v)
  _Float16* Hc = (_Float16*)(ws + 48 * MB);   // [T,NH]  f16   4MB (chain h)
  float*    Fx = (float*)   (ws + 52 * MB);   // [T] softplus part of F(x)
  float*    Fv = (float*)   (ws + 52 * MB + 8192);

  init_zero<<<dim3((T_DIM + 255) / 256), dim3(256), 0, stream>>>(Fx, Fv, out);

  convert_transpose_w<<<dim3(NV_DIM / 32, NH_DIM / 32), dim3(32, 8), 0, stream>>>(
      w, Wf, WT);

  const int nxe = T_DIM * NV_DIM;
  convert_x<<<dim3((nxe + 255) / 256), dim3(256), 0, stream>>>(x, Vx, Vc, nxe);

  const dim3 blk(128);
  const dim3 gridH(T_DIM / 128, NH_DIM / 64);   // (8, 32)
  const dim3 gridV(T_DIM / 128, NV_DIM / 64);   // (8, 64)

  for (int k = 0; k < K_STEPS; ++k) {
    // H = sample(sigmoid(Vc @ Wf^T + bh))
    rbm_gemm_nt<<<gridH, blk, 0, stream>>>(Vc, Wf, bh, Hc, (float*)nullptr,
                                           NV_DIM, NH_DIM, 0,
                                           0xC0FFEEu + 2u * (unsigned)k);
    // V = sample(sigmoid(Hc @ WT^T + bv))
    rbm_gemm_nt<<<gridV, blk, 0, stream>>>(Hc, WT, bv, Vc, (float*)nullptr,
                                           NH_DIM, NV_DIM, 0,
                                           0xC0FFEEu + 2u * (unsigned)k + 1u);
  }

  // free-energy softplus terms for x and v_o (hidden-unit GEMM shape)
  rbm_gemm_nt<<<gridH, blk, 0, stream>>>(Vx, Wf, bh, (_Float16*)nullptr, Fx,
                                         NV_DIM, NH_DIM, 1, 0u);
  rbm_gemm_nt<<<gridH, blk, 0, stream>>>(Vc, Wf, bh, (_Float16*)nullptr, Fv,
                                         NV_DIM, NH_DIM, 1, 0u);

  rbm_finalize<<<dim3(T_DIM), dim3(256), 0, stream>>>(x, Vc, bv, Fx, Fv, out);

  (void)in_sizes; (void)n_in; (void)out_size; (void)ws_size;
}
